// MultiHeadAttention_12756052869526
// MI455X (gfx1250) — matrix operand order repaired and checked by reading, not compiled, not run on hardware
//
#include <hip/hip_runtime.h>
#include <cstdint>

// ---------------------------------------------------------------------------
// MHA layer for MI455X (gfx1250, wave32, WMMA f16->f32).
//   D_MODEL=1024, H=16, DEPTH=64, B=2, S=2048
// Pipeline:
//   K0: one-shot f32->f16 conversion of x_q/x_k/x_v and Wq/Wk/Wv/Wo
//   K1: qkv projection via async-to-LDS double-buffered staging + WMMA
//       (Q,K staged out as [B,H,S,D] f16; V transposed to [B,H,D,S] f16)
//   K2: per (b,h,32-query tile): logits WMMA -> 256KB LDS f32 tile ->
//       softmax (exp kept unnormalized in LDS, 1/sum folded into ctx acc) ->
//       nontemporal f32 attn write -> ctx WMMA -> ctx f16
//   K3: out = LN(ctx @ Wo16^T + bo + residual), fused via 128KB LDS row tile
// ---------------------------------------------------------------------------

#define D_MODEL 1024
#define N_HEADS 16
#define DEPTH   64
#define BATCH   2
#define SEQ     2048

typedef __attribute__((ext_vector_type(16))) _Float16 v16h;
typedef __attribute__((ext_vector_type(8)))  _Float16 v8h;
typedef __attribute__((ext_vector_type(8)))  float    v8f;
typedef __attribute__((ext_vector_type(4)))  float    v4f;

__device__ __forceinline__ v16h cat8(v8h lo, v8h hi) {
  return __builtin_shufflevector(lo, hi, 0,1,2,3,4,5,6,7,8,9,10,11,12,13,14,15);
}

// A fragment (16x32 f16, ISA 7.12.2): lane holds row m=(lane&15);
// halfs at k = h8 + {0..7} and k = h8 + 16 + {0..7}, h8 = (lane>>4)*8.
// Caller passes p = &src[row*ld + kbase + h8].
__device__ __forceinline__ v16h load_a_h16(const _Float16* p) {
  return cat8(*(const v8h*)p, *(const v8h*)(p + 16));
}
// B fragment (32x16 f16): lane holds col n=(lane&15) = a source row;
// k = h8 + {0..7} and h8 + 16 + {0..7}, h8 = (lane>>4)*8. Caller passes p = &src[n*ld + kbase + h8].
__device__ __forceinline__ v16h load_b_h16(const _Float16* p) {
  return cat8(*(const v8h*)p, *(const v8h*)(p + 16));
}

__device__ __forceinline__ v8f wmma16(v16h a, v16h b, v8f c) {
  return __builtin_amdgcn_wmma_f32_16x16x32_f16(
      /*neg_a=*/false, a, /*neg_b=*/false, b,
      /*c_mod=*/(short)0, c, /*reuse_a=*/false, /*reuse_b=*/false);
}

// Async global->LDS copy, 16B per lane (512B per wave-instruction).
__device__ __forceinline__ void async_cp16(uint32_t lds_byte_addr,
                                           const _Float16* gsrc) {
  asm volatile("global_load_async_to_lds_b128 %0, %1, off"
               :: "v"(lds_byte_addr), "v"(gsrc) : "memory");
}
__device__ __forceinline__ void wait_async0() {
#if __has_builtin(__builtin_amdgcn_s_wait_asynccnt)
  __builtin_amdgcn_s_wait_asynccnt(0);
#else
  asm volatile("s_wait_asynccnt 0x0" ::: "memory");
#endif
}

// ---------------------------------------------------------------------------
// Kernel 0: one-shot f32 -> f16 conversion (8 elements / thread).
// ---------------------------------------------------------------------------
__global__ __launch_bounds__(256) void cvt_f16_kernel(
    const float* __restrict__ src, _Float16* __restrict__ dst, int n) {
  const int i = (blockIdx.x * 256 + threadIdx.x) * 8;
  if (i + 8 <= n) {
    v4f a = *(const v4f*)(src + i);
    v4f b = *(const v4f*)(src + i + 4);
    v8h o;
#pragma unroll
    for (int j = 0; j < 4; ++j) {
      o[j]     = (_Float16)a[j];
      o[4 + j] = (_Float16)b[j];
    }
    *(v8h*)(dst + i) = o;
  }
}

// ---------------------------------------------------------------------------
// Kernel 1: out = X16 @ W16^T + bias with async-LDS double buffering.
// Block = 8 waves, tile 64 rows x 128 cols; wave = 32x32 (2x2 WMMA tiles).
// Per 32-k chunk: stage x-tile (64x32 f16 = 4KB) + w-tile (128x32 f16 = 8KB)
// via 3 global_load_async_to_lds_b128 per wave. LDS = 2 x 12KB.
// vmode==0 -> [B,H,S,D]; vmode==1 -> [B,H,D,S] (transposed for V).
// ---------------------------------------------------------------------------
#define K1_BUF_H (2048 + 4096)   // halfs per buffer: xs[64][32] + ws[128][32]

__device__ __forceinline__ void k1_stage(const _Float16* __restrict__ x16,
                                         const _Float16* __restrict__ w16,
                                         int rbase0, int cbase0, int kb,
                                         _Float16* ldsbuf, int wv, int lane) {
#pragma unroll
  for (int j = 0; j < 3; ++j) {
    const int g = (wv * 3 + j) * 32 + lane;   // 0..767 flat 16B-block index
    const _Float16* gp;
    _Float16* lp;
    if (g < 256) {                            // x tile: 256 x 16B = 4KB
      const int row = g >> 2, seg = g & 3;
      gp = x16 + (size_t)(rbase0 + row) * D_MODEL + kb + seg * 8;
      lp = ldsbuf + g * 8;
    } else {                                  // w tile: 512 x 16B = 8KB
      const int gw = g - 256, row = gw >> 2, seg = gw & 3;
      gp = w16 + (size_t)(cbase0 + row) * D_MODEL + kb + seg * 8;
      lp = ldsbuf + 2048 + gw * 8;
    }
    async_cp16((uint32_t)(uintptr_t)lp, gp);
  }
}

__global__ __launch_bounds__(256) void qkv_proj_kernel(
    const _Float16* __restrict__ x16, const _Float16* __restrict__ w16,
    const float* __restrict__ bias, _Float16* __restrict__ out, int vmode) {
  __shared__ _Float16 lds[2 * K1_BUF_H];     // 24 KB
  const int tid  = threadIdx.x;
  const int lane = tid & 31, wv = tid >> 5;
  const int h16  = lane >> 4, ln = lane & 15;
  const int rw   = wv >> 2, cw = wv & 3;
  const int rbase0 = blockIdx.y * 64;
  const int cbase0 = blockIdx.x * 128;

  v8f acc[2][2] = {};
  k1_stage(x16, w16, rbase0, cbase0, 0, lds, wv, lane);

  for (int kc = 0; kc < D_MODEL / 32; ++kc) {
    wait_async0();
    __syncthreads();
    const int buf = kc & 1;
    if (kc + 1 < D_MODEL / 32)
      k1_stage(x16, w16, rbase0, cbase0, (kc + 1) * 32,
               lds + (1 - buf) * K1_BUF_H, wv, lane);

    const _Float16* xs  = lds + buf * K1_BUF_H;
    const _Float16* wsm = xs + 2048;
    v16h a0 = load_a_h16(xs  + (rw * 32 + ln) * 32       + h16 * 8);
    v16h a1 = load_a_h16(xs  + (rw * 32 + 16 + ln) * 32  + h16 * 8);
    v16h b0 = load_b_h16(wsm + (cw * 32 + ln) * 32       + h16 * 8);
    v16h b1 = load_b_h16(wsm + (cw * 32 + 16 + ln) * 32  + h16 * 8);
    acc[0][0] = wmma16(a0, b0, acc[0][0]);
    acc[0][1] = wmma16(a0, b1, acc[0][1]);
    acc[1][0] = wmma16(a1, b0, acc[1][0]);
    acc[1][1] = wmma16(a1, b1, acc[1][1]);
  }

  const int rbase = rbase0 + rw * 32;
  const int cbase = cbase0 + cw * 32;
  const float bv[2] = { bias[cbase + ln], bias[cbase + 16 + ln] };
#pragma unroll
  for (int ri = 0; ri < 2; ++ri)
#pragma unroll
    for (int ci = 0; ci < 2; ++ci)
#pragma unroll
      for (int e = 0; e < 8; ++e) {
        const int row = rbase + ri * 16 + e + h16 * 8;  // C: M = e + (lane>>4)*8
        const int col = cbase + ci * 16 + ln;           // C: N = lane&15
        const int b = row >> 11, s = row & (SEQ - 1);
        const int hh = col >> 6, d = col & (DEPTH - 1);
        const size_t idx = vmode
            ? ((size_t)((b * N_HEADS + hh) * DEPTH + d)) * SEQ + s
            : ((size_t)((b * N_HEADS + hh) * SEQ + s)) * DEPTH + d;
        out[idx] = (_Float16)(acc[ri][ci][e] + bv[ci]);
      }
}

// ---------------------------------------------------------------------------
// Kernel 2: attention. One block per (b, h, 32-query tile). 2048 blocks.
// LDS: 32x2048 f32 score tile (256 KB, fits 320 KB WGP LDS) + 544f scratch.
// ---------------------------------------------------------------------------
__global__ __launch_bounds__(256) void attn_kernel(
    const _Float16* __restrict__ qh, const _Float16* __restrict__ kh,
    const _Float16* __restrict__ vht, const float* __restrict__ mask,
    float* __restrict__ attn_out, _Float16* __restrict__ ctx) {
  extern __shared__ float smem[];           // [32][SEQ] logits -> exp values
  float* red = smem + 32 * SEQ;             // [544]: 2x256 partials + 32 inv

  const int tid  = threadIdx.x;
  const int lane = tid & 31, wv = tid >> 5;
  const int h16  = lane >> 4, ln = lane & 15;
  const int bid  = blockIdx.x;
  const int qt   = bid & 63;                // 64 query tiles per (b,h)
  const int bh   = bid >> 6;                // b*16 + h
  const int b    = bh >> 4;
  const int qbase = qt * 32;

  const _Float16* qb = qh + (size_t)bh * SEQ * DEPTH;
  const _Float16* kb = kh + (size_t)bh * SEQ * DEPTH;

  // ---- phase 1: logits = (Q K^T) * 1/sqrt(64) + mask * -1e9 --------------
  v16h aA[2][2];
#pragma unroll
  for (int qi = 0; qi < 2; ++qi)
#pragma unroll
    for (int c = 0; c < 2; ++c)
      aA[qi][c] = load_a_h16(qb + (size_t)(qbase + qi * 16 + ln) * DEPTH +
                             c * 32 + h16 * 8);

  for (int kt = wv; kt < SEQ / 16; kt += 8) {
    const _Float16* kp = kb + (size_t)(kt * 16 + ln) * DEPTH + h16 * 8;
    v16h b0 = load_b_h16(kp);
    v16h b1 = load_b_h16(kp + 32);
    v8f acc0 = {}, acc1 = {};
    acc0 = wmma16(aA[0][0], b0, acc0);
    acc0 = wmma16(aA[0][1], b1, acc0);
    acc1 = wmma16(aA[1][0], b0, acc1);
    acc1 = wmma16(aA[1][1], b1, acc1);
    const int kcol = kt * 16 + ln;
    const float mk = mask[b * SEQ + kcol] * -1e9f;
#pragma unroll
    for (int e = 0; e < 8; ++e) {
      smem[(e + h16 * 8) * SEQ + kcol]      = acc0[e] * 0.125f + mk;
      smem[(16 + e + h16 * 8) * SEQ + kcol] = acc1[e] * 0.125f + mk;
    }
  }
  __syncthreads();

  // ---- phase 2: softmax over SEQ, 8 lanes per row ------------------------
  const int row = tid >> 3, sub = tid & 7;
  float* rp = smem + row * SEQ + sub * 256;
  float m = -3.0e38f;
  for (int i = 0; i < 256; ++i) m = fmaxf(m, rp[i]);
  red[row * 8 + sub] = m;
  __syncthreads();
  float rm = red[row * 8];
#pragma unroll
  for (int j = 1; j < 8; ++j) rm = fmaxf(rm, red[row * 8 + j]);
  float s = 0.0f;
  for (int i = 0; i < 256; ++i) {
    float e = __expf(rp[i] - rm);
    rp[i] = e;                       // keep UNNORMALIZED exp in LDS
    s += e;
  }
  red[256 + row * 8 + sub] = s;
  __syncthreads();
  float rs = 0.0f;
#pragma unroll
  for (int j = 0; j < 8; ++j) rs += red[256 + row * 8 + j];
  const float inv = 1.0f / rs;
  if (sub == 0) red[512 + row] = inv;  // folded into ctx accumulator later
  // dominant HBM write: streamed, never re-read -> nontemporal
  float* ao = attn_out + ((size_t)bh * SEQ + qbase + row) * SEQ + sub * 256;
  for (int i = 0; i < 256; ++i)
    __builtin_nontemporal_store(rp[i] * inv, ao + i);
  __syncthreads();

  // ---- phase 3: ctx = softmax(QK^T) @ V; A from LDS exp values -----------
  const int qi = wv >> 2, nt = wv & 3;      // 2 q-subtiles x 4 d-subtiles
  const float* arow = smem + (qi * 16 + ln) * SEQ + h16 * 8;
  const _Float16* brow = vht + ((size_t)bh * DEPTH + nt * 16 + ln) * SEQ + h16 * 8;
  v8f acc = {};
  for (int kbo = 0; kbo < SEQ; kbo += 32) {
    const float* p = arow + kbo;
    v16h a;
#pragma unroll
    for (int i = 0; i < 8; ++i) {
      a[i]     = (_Float16)p[i];
      a[8 + i] = (_Float16)p[16 + i];
    }
    acc = wmma16(a, load_b_h16(brow + kbo), acc);
  }
  const int hcol = (bh & 15) * DEPTH + nt * 16 + ln;   // merged-head column
#pragma unroll
  for (int e = 0; e < 8; ++e) {
    const float invm = red[512 + qi * 16 + e + h16 * 8];  // per-row 1/sum
    const int rowg = b * SEQ + qbase + qi * 16 + e + h16 * 8;
    ctx[(size_t)rowg * D_MODEL + hcol] = (_Float16)(acc[e] * invm);
  }
}

// ---------------------------------------------------------------------------
// Kernel 3: out = LayerNorm(ctx @ Wo16^T + bo + residual). 32 rows / block,
// full 1024-wide f32 row tile in 128 KB LDS so `out` is written once.
// ---------------------------------------------------------------------------
__global__ __launch_bounds__(256) void oproj_ln_kernel(
    const _Float16* __restrict__ ctx, const _Float16* __restrict__ wo16,
    const float* __restrict__ wob, const float* __restrict__ resid,
    const float* __restrict__ g, const float* __restrict__ beta,
    float* __restrict__ out) {
  extern __shared__ float smem[];           // [32][D_MODEL]
  float* red = smem + 32 * D_MODEL;         // [512]

  const int tid  = threadIdx.x;
  const int lane = tid & 31, wv = tid >> 5;
  const int h16  = lane >> 4, ln = lane & 15;
  const int rbase = blockIdx.x * 32;

#pragma unroll 1
  for (int t = 0; t < 8; ++t) {
    const int nt = wv + t * 8;              // 64 col tiles of 16
    const _Float16* brow = wo16 + (size_t)(nt * 16 + ln) * D_MODEL + h16 * 8;
    const float bv = wob[nt * 16 + ln];
#pragma unroll 1
    for (int qi = 0; qi < 2; ++qi) {
      const _Float16* arow = ctx + (size_t)(rbase + qi * 16 + ln) * D_MODEL + h16 * 8;
      v8f acc = {};
      for (int kb = 0; kb < D_MODEL; kb += 32)
        acc = wmma16(load_a_h16(arow + kb), load_b_h16(brow + kb), acc);
#pragma unroll
      for (int e = 0; e < 8; ++e)
        smem[(qi * 16 + e + h16 * 8) * D_MODEL + nt * 16 + ln] = acc[e] + bv;
    }
  }
  __syncthreads();

  // residual + layernorm: 8 lanes per row, 128 elements each
  const int row = tid >> 3, sub = tid & 7;
  float* rp = smem + row * D_MODEL + sub * 128;
  const float* rr = resid + (size_t)(rbase + row) * D_MODEL + sub * 128;
  float sm = 0.0f, sq = 0.0f;
  for (int i = 0; i < 128; ++i) {
    float x = rp[i] + rr[i];
    rp[i] = x;
    sm += x; sq += x * x;
  }
  red[row * 8 + sub] = sm;
  red[256 + row * 8 + sub] = sq;
  __syncthreads();
  float ts = 0.0f, tq = 0.0f;
#pragma unroll
  for (int j = 0; j < 8; ++j) { ts += red[row * 8 + j]; tq += red[256 + row * 8 + j]; }
  const float mu   = ts * (1.0f / D_MODEL);
  const float var  = tq * (1.0f / D_MODEL) - mu * mu;
  const float rstd = rsqrtf(var + 1e-5f);
  float* op = out + (size_t)(rbase + row) * D_MODEL + sub * 128;
  const float* gp = g    + sub * 128;
  const float* bp = beta + sub * 128;
  for (int i = 0; i < 128; ++i)
    op[i] = (rp[i] - mu) * rstd * gp[i] + bp[i];
}

// ---------------------------------------------------------------------------
extern "C" void kernel_launch(void* const* d_in, const int* in_sizes, int n_in,
                              void* d_out, int out_size, void* d_ws, size_t ws_size,
                              hipStream_t stream) {
  (void)in_sizes; (void)n_in; (void)out_size; (void)ws_size;
  const float* q    = (const float*)d_in[0];
  const float* k    = (const float*)d_in[1];
  const float* v    = (const float*)d_in[2];
  const float* mask = (const float*)d_in[3];
  const float* wq_w = (const float*)d_in[4];
  const float* wq_b = (const float*)d_in[5];
  const float* wk_w = (const float*)d_in[6];
  const float* wk_b = (const float*)d_in[7];
  const float* wv_w = (const float*)d_in[8];
  const float* wv_b = (const float*)d_in[9];
  const float* wo_w = (const float*)d_in[10];
  const float* wo_b = (const float*)d_in[11];
  const float* ln_g = (const float*)d_in[12];
  const float* ln_b = (const float*)d_in[13];

  float* out0 = (float*)d_out;
  float* attn = out0 + (size_t)BATCH * SEQ * D_MODEL;   // tuple order (out, attn)

  const size_t tszA = (size_t)BATCH * SEQ * D_MODEL;    // 4M elements
  const size_t tszW = (size_t)D_MODEL * D_MODEL;        // 1M elements
  _Float16* x16q = (_Float16*)d_ws;                     // f16 inputs
  _Float16* x16k = x16q + tszA;
  _Float16* x16v = x16k + tszA;
  _Float16* w16q = x16v + tszA;                         // f16 weights
  _Float16* w16k = w16q + tszW;
  _Float16* w16v = w16k + tszW;
  _Float16* w16o = w16v + tszW;
  _Float16* qh   = w16o + tszW;                         // [B,H,S,D]
  _Float16* kh   = qh + tszA;                           // [B,H,S,D]
  _Float16* vht  = kh + tszA;                           // [B,H,D,S]
  _Float16* ctx  = vht + tszA;                          // [B*S, D_MODEL]

  const dim3 blk(256);
  const int gA = (int)(tszA / 2048), gW = (int)(tszW / 2048);
  cvt_f16_kernel<<<gA, blk, 0, stream>>>(q,    x16q, (int)tszA);
  cvt_f16_kernel<<<gA, blk, 0, stream>>>(k,    x16k, (int)tszA);
  cvt_f16_kernel<<<gA, blk, 0, stream>>>(v,    x16v, (int)tszA);
  cvt_f16_kernel<<<gW, blk, 0, stream>>>(wq_w, w16q, (int)tszW);
  cvt_f16_kernel<<<gW, blk, 0, stream>>>(wk_w, w16k, (int)tszW);
  cvt_f16_kernel<<<gW, blk, 0, stream>>>(wv_w, w16v, (int)tszW);
  cvt_f16_kernel<<<gW, blk, 0, stream>>>(wo_w, w16o, (int)tszW);

  qkv_proj_kernel<<<dim3(8, 64), blk, 0, stream>>>(x16q, w16q, wq_b, qh, 0);
  qkv_proj_kernel<<<dim3(8, 64), blk, 0, stream>>>(x16k, w16k, wk_b, kh, 0);
  qkv_proj_kernel<<<dim3(8, 64), blk, 0, stream>>>(x16v, w16v, wv_b, vht, 1);

  const size_t sm2 = (size_t)32 * SEQ * sizeof(float) + 544 * sizeof(float);
  attn_kernel<<<dim3(BATCH * N_HEADS * (SEQ / 32)), blk, sm2, stream>>>(
      qh, kh, vht, mask, attn, ctx);

  const size_t sm3 = (size_t)32 * D_MODEL * sizeof(float) + 512 * sizeof(float);
  oproj_ln_kernel<<<dim3((BATCH * SEQ) / 32), blk, sm3, stream>>>(
      ctx, w16o, wo_b, q, ln_g, ln_b, out0);
}
